// ScaledDotProductAttention_45217415692657
// MI455X (gfx1250) — matrix operand order repaired and checked by reading, not compiled, not run on hardware
//
#include <hip/hip_runtime.h>

// ---- problem constants ----
#define F_SIZE 24
#define SEQ    577            // S
#define NGRID  576            // N
#define PDIM   47             // 2F-1
#define NP     2209           // 47*47 relative positions
#define NP16   2224           // NP padded to 16 (N-dim tiles in kernel 1)
#define NP32   2240           // NP padded to 32 (K-dim steps in kernel 2)
#define HD     64             // head dim
#define KPAD   608            // 19*32, S padded to 32 for K-steps
#define LPITCH 592            // 37*16, logits row pitch
#define OUT_ELEMS  ((size_t)16*16*SEQ*HD)        // 9,453,568
#define ATTN_ELEMS ((size_t)16*16*SEQ*SEQ)       // 85,229,824

typedef __attribute__((ext_vector_type(16))) _Float16 v16h;
typedef __attribute__((ext_vector_type(8)))  float    v8f;

__device__ __forceinline__ v8f wmma16(v16h a, v16h b, v8f c) {
  // D = A(16x32 f16) x B(32x16 f16) + C(16x16 f32)
  return __builtin_amdgcn_wmma_f32_16x16x32_f16(false, a, false, b,
                                                (short)0, c, false, false);
}

// ---------------------------------------------------------------------------
// Kernel 0a: f16 copies of the position fmaps.
//   kf  [NP16][64]  row-major (B-frags of kernel1: N=p, K=d)
//   vfT [64][NP32]  transposed (B-frags of kernel2: K=p, N=d)
// ---------------------------------------------------------------------------
__global__ __launch_bounds__(256)
void k0_fmaps(const float* __restrict__ kf_in, const float* __restrict__ vf_in,
              _Float16* __restrict__ kf, _Float16* __restrict__ vfT)
{
  int idx = blockIdx.x * 256 + threadIdx.x;     // over NP32*64
  if (idx >= NP32 * HD) return;
  int p = idx >> 6, d = idx & 63;
  float kv = (p < NP) ? kf_in[(size_t)p * HD + d] : 0.f;
  float vv = (p < NP) ? vf_in[(size_t)p * HD + d] : 0.f;
  if (p < NP16) kf[(size_t)p * HD + d] = (_Float16)kv;
  vfT[(size_t)d * NP32 + p] = (_Float16)vv;
}

// ---------------------------------------------------------------------------
// Kernel 0b: vT[bh][d][k] f16 transposed copy of v (zero-padded k to KPAD)
// ---------------------------------------------------------------------------
__global__ __launch_bounds__(256)
void k0_vt(const float* __restrict__ v, _Float16* __restrict__ vT)
{
  size_t idx = (size_t)blockIdx.x * 256 + threadIdx.x;
  size_t total = (size_t)256 * HD * KPAD;
  if (idx >= total) return;
  int kk = (int)(idx % KPAD);
  size_t t = idx / KPAD;
  int d  = (int)(t % HD);
  int bh = (int)(t / HD);
  float val = (kk < SEQ) ? v[((size_t)bh * SEQ + kk) * HD + d] : 0.f;
  vT[idx] = (_Float16)val;
}

// ---------------------------------------------------------------------------
// Kernel 1: scores + softmax -> attn.  grid = (37 q-tiles, 256 bh), 256 thr.
// ---------------------------------------------------------------------------
__global__ __launch_bounds__(256)
void k1_scores(const float* __restrict__ q, const float* __restrict__ kmat,
               const float* __restrict__ kgem, const _Float16* __restrict__ kf,
               float* __restrict__ attn)
{
  __shared__ _Float16 qs16[16 * HD];          // scaled q tile, f16
  __shared__ _Float16 W[16 * NP16];           // qs @ fmap^T  (f16)
  __shared__ float    logits[16 * LPITCH];
  __shared__ float    red[16 * 16];

  const int qt   = blockIdx.x;                // 0..36
  const int bh   = blockIdx.y;                // 0..255
  const int tid  = threadIdx.x;
  const int lane = tid & 31;
  const int wave = tid >> 5;                  // 0..7
  const size_t qofs = (size_t)bh * SEQ * HD;

  // Phase A: stage scaled q tile (rows past S -> 0)
  for (int e = tid; e < 16 * HD; e += 256) {
    int r = e >> 6, d = e & 63;
    int qg = qt * 16 + r;
    float val = (qg < SEQ) ? q[qofs + (size_t)qg * HD + d] * 0.125f : 0.f;
    qs16[e] = (_Float16)val;
  }
  __syncthreads();

  // A fragments (shared by all WMMAs in this block): contiguous K halves (k = 16h + s)
  const int arow  = lane & 15;
  const int khalf = (lane >> 4) ? 16 : 0;
  const int bcol  = lane & 15;
  const int bhi   = (lane >> 4) ? 16 : 0;
  v16h afrag[2];
  for (int f = 0; f < 2; ++f) {
    int dbase = f * 32;
    for (int j = 0; j < 8; ++j) {
      afrag[f][j]     = qs16[arow * HD + dbase + khalf + j];
      afrag[f][j + 8] = qs16[arow * HD + dbase + khalf + 8 + j];
    }
  }
  const int mbase = (lane >> 4) ? 8 : 0;

  // Phase B: W = qs @ kf^T over 139 p-tiles (skip for the gem-query tile)
  if (qt != 36) {
    for (int pt = wave; pt < 139; pt += 8) {
      v8f acc = {};
      for (int f = 0; f < 2; ++f) {
        v16h b;
        const _Float16* src = kf + (size_t)(pt * 16 + bcol) * HD + f * 32 + bhi;
        for (int j = 0; j < 16; ++j) b[j] = src[j];
        acc = wmma16(afrag[f], b, acc);
      }
      for (int r = 0; r < 8; ++r)
        W[(mbase + r) * NP16 + pt * 16 + bcol] = (_Float16)acc[r];
    }
  }

  // Phase C: content scores qs @ k^T (f32 k converted on the fly)
  for (int kt = wave; kt < 37; kt += 8) {
    v8f acc = {};
    int krow = kt * 16 + bcol;
    bool valid = krow < SEQ;
    for (int f = 0; f < 2; ++f) {
      v16h b;
      if (valid) {
        const float* src = kmat + qofs + (size_t)krow * HD + f * 32 + bhi;
        for (int j = 0; j < 16; ++j) b[j] = (_Float16)src[j];
      } else {
        for (int j = 0; j < 16; ++j) b[j] = (_Float16)0.f;
      }
      acc = wmma16(afrag[f], b, acc);
    }
    for (int r = 0; r < 8; ++r)
      logits[(mbase + r) * LPITCH + kt * 16 + bcol] = acc[r];
  }
  __syncthreads();

  // Phase D: add relative-position bias (gather from W) + gem terms
  if (qt != 36) {
    for (int e = tid; e < 16 * SEQ; e += 256) {
      int r = e / SEQ, kk = e % SEQ;
      int qg = qt * 16 + r;                     // always < 576 here
      float add;
      if (kk < NGRID) {
        int dy = qg / F_SIZE - kk / F_SIZE + (F_SIZE - 1);
        int dx = qg % F_SIZE - kk % F_SIZE + (F_SIZE - 1);
        add = (float)W[r * NP16 + dy * PDIM + dx];
      } else {                                  // k = gem column
        float s = 0.f;
        for (int d = 0; d < HD; ++d)
          s += (float)qs16[r * HD + d] * kgem[(size_t)qg * HD + d];
        add = s;
      }
      logits[r * LPITCH + kk] += add;
    }
  } else {
    // gem query row (q = 576): pos[576,k] = kgem[k+1] (k<576), kgem[576] else
    for (int kk = tid; kk < SEQ; kk += 256) {
      int gi = (kk < NGRID) ? kk + 1 : NGRID;
      float s = 0.f;
      for (int d = 0; d < HD; ++d)
        s += (float)qs16[d] * kgem[(size_t)gi * HD + d];
      logits[kk] += s;
    }
  }
  __syncthreads();

  // Phase E: softmax per row (16 rows x 16 lanes each)
  {
    int r = tid >> 4, l = tid & 15;
    float m = -1e30f;
    for (int kk = l; kk < SEQ; kk += 16) m = fmaxf(m, logits[r * LPITCH + kk]);
    red[r * 16 + l] = m;
    __syncthreads();
    m = red[r * 16];
    for (int i = 1; i < 16; ++i) m = fmaxf(m, red[r * 16 + i]);
    float s = 0.f;
    for (int kk = l; kk < SEQ; kk += 16) {
      float ev = __expf(logits[r * LPITCH + kk] - m);
      logits[r * LPITCH + kk] = ev;
      s += ev;
    }
    __syncthreads();
    red[r * 16 + l] = s;
    __syncthreads();
    s = 0.f;
    for (int i = 0; i < 16; ++i) s += red[r * 16 + i];
    float inv = 1.f / s;
    for (int kk = l; kk < SEQ; kk += 16) logits[r * LPITCH + kk] *= inv;
  }
  __syncthreads();

  // Phase F: write attn
  for (int e = tid; e < 16 * SEQ; e += 256) {
    int r = e / SEQ, kk = e % SEQ;
    int qg = qt * 16 + r;
    if (qg < SEQ)
      attn[((size_t)bh * SEQ + qg) * SEQ + kk] = logits[r * LPITCH + kk];
  }
}

// ---------------------------------------------------------------------------
// Kernel 2: out = attn@v + pos-value term.  grid = (37, 256), 256 threads.
// ---------------------------------------------------------------------------
__global__ __launch_bounds__(256)
void k2_out(const float* __restrict__ vgem, const _Float16* __restrict__ vfT,
            const _Float16* __restrict__ vT, const float* __restrict__ attn,
            float* __restrict__ out)
{
  __shared__ _Float16 ah[16 * NP32];          // relative-offset histogram
  __shared__ _Float16 at16[16 * KPAD];        // attn tile, f16
  __shared__ float    partial[8 * 16 * HD];   // per-wave partial outputs
  __shared__ float    gcol[16];               // attn[:,576] in f32
  __shared__ float    lastrow[4 * HD];        // gem-query pos-value partials

  const int qt   = blockIdx.x;
  const int bh   = blockIdx.y;
  const int tid  = threadIdx.x;
  const int lane = tid & 31;
  const int wave = tid >> 5;

  // zero A_hist (dword stores)
  for (int e = tid; e < 16 * NP32 / 2; e += 256)
    ((unsigned int*)ah)[e] = 0u;
  // stage attn tile (f32 -> f16, zero-padded), capture gem column
  for (int e = tid; e < 16 * KPAD; e += 256) {
    int r = e / KPAD, kk = e % KPAD;
    int qg = qt * 16 + r;
    float a = 0.f;
    if (qg < SEQ && kk < SEQ) a = attn[((size_t)bh * SEQ + qg) * SEQ + kk];
    at16[e] = (_Float16)a;
    if (kk == NGRID) gcol[r] = a;
  }
  __syncthreads();

  // scatter attn -> A_hist (injective per q; grid q-tiles only)
  if (qt != 36) {
    for (int e = tid; e < 16 * NGRID; e += 256) {
      int r = e / NGRID, kk = e % NGRID;
      int qg = qt * 16 + r;                   // < 576 here
      int dy = qg / F_SIZE - kk / F_SIZE + (F_SIZE - 1);
      int dx = qg % F_SIZE - kk % F_SIZE + (F_SIZE - 1);
      ah[r * NP32 + dy * PDIM + dx] = at16[r * KPAD + kk];
    }
  }
  __syncthreads();

  const int arow  = lane & 15;
  const int khalf = (lane >> 4) ? 16 : 0;
  const int bcol  = lane & 15;
  const int bhi   = (lane >> 4) ? 16 : 0;
  const int mbase = (lane >> 4) ? 8 : 0;
  v8f acc[4] = {v8f{}, v8f{}, v8f{}, v8f{}};

  // content term: attn @ v  (19 K-steps of 32, split across 8 waves)
  const _Float16* vtb = vT + (size_t)bh * HD * KPAD;
  for (int ks = wave; ks < 19; ks += 8) {
    int kb = ks * 32;
    v16h a;
    for (int j = 0; j < 8; ++j) {
      a[j]     = at16[arow * KPAD + kb + khalf + j];
      a[j + 8] = at16[arow * KPAD + kb + khalf + 8 + j];
    }
    for (int nt = 0; nt < 4; ++nt) {
      v16h b;
      const _Float16* src = vtb + (size_t)(nt * 16 + bcol) * KPAD + kb + bhi;
      for (int j = 0; j < 16; ++j) b[j] = src[j];
      acc[nt] = wmma16(a, b, acc[nt]);
    }
  }
  // pos-value term: A_hist @ v_fmap (70 K-steps of 32)
  if (qt != 36) {
    for (int ps = wave; ps < 70; ps += 8) {
      int pb = ps * 32;
      v16h a;
      for (int j = 0; j < 8; ++j) {
        a[j]     = ah[arow * NP32 + pb + khalf + j];
        a[j + 8] = ah[arow * NP32 + pb + khalf + 8 + j];
      }
      for (int nt = 0; nt < 4; ++nt) {
        v16h b;
        const _Float16* src = vfT + (size_t)(nt * 16 + bcol) * NP32 + pb + bhi;
        for (int j = 0; j < 16; ++j) b[j] = src[j];
        acc[nt] = wmma16(a, b, acc[nt]);
      }
    }
  }
  // dump per-wave partials (D layout: N=lane&15, M=r+mbase)
  for (int nt = 0; nt < 4; ++nt)
    for (int r = 0; r < 8; ++r)
      partial[(wave * 16 + mbase + r) * HD + nt * 16 + (lane & 15)] = acc[nt][r];
  __syncthreads();

  // gem-query pos-value term (q = 576): sum_k attn[576,k] * vgem[kg(k)]
  if (qt == 36) {
    int d = tid & 63, g = tid >> 6;           // 4 k-groups
    float s = 0.f;
    for (int kk = g; kk < SEQ; kk += 4) {
      int gi = (kk < NGRID) ? kk + 1 : NGRID;
      s += attn[((size_t)bh * SEQ + NGRID) * SEQ + kk] * vgem[(size_t)gi * HD + d];
    }
    lastrow[g * HD + d] = s;
  }
  __syncthreads();

  // combine partials + gem terms, store
  for (int e = tid; e < 16 * HD; e += 256) {
    int r = e >> 6, d = e & 63;
    int qg = qt * 16 + r;
    if (qg >= SEQ) continue;
    float s = 0.f;
    for (int w = 0; w < 8; ++w) s += partial[(w * 16 + r) * HD + d];
    if (qg < NGRID) {
      s += gcol[r] * vgem[(size_t)qg * HD + d];   // v_pos[q,576] = vgem[q]
    } else {
      s += lastrow[d] + lastrow[HD + d] + lastrow[2 * HD + d] + lastrow[3 * HD + d];
    }
    out[((size_t)bh * SEQ + qg) * HD + d] = s;
  }
}

// ---------------------------------------------------------------------------
extern "C" void kernel_launch(void* const* d_in, const int* in_sizes, int n_in,
                              void* d_out, int out_size, void* d_ws, size_t ws_size,
                              hipStream_t stream) {
  (void)in_sizes; (void)n_in; (void)out_size; (void)ws_size;
  const float* q     = (const float*)d_in[0];
  const float* k     = (const float*)d_in[1];
  const float* v     = (const float*)d_in[2];
  const float* kgem  = (const float*)d_in[3];
  const float* kfmap = (const float*)d_in[4];
  const float* vgem  = (const float*)d_in[5];
  const float* vfmap = (const float*)d_in[6];

  float* out_ptr  = (float*)d_out;
  float* attn_ptr = (float*)d_out + OUT_ELEMS;

  // workspace layout (~20.5 MB)
  _Float16* kf  = (_Float16*)d_ws;                                    // NP16*64
  _Float16* vfT = (_Float16*)((char*)d_ws + (size_t)NP16 * HD * 2);   // 64*NP32
  _Float16* vT  = (_Float16*)((char*)d_ws + (size_t)NP16 * HD * 2
                                          + (size_t)HD * NP32 * 2);   // 256*64*KPAD

  int g0 = (NP32 * HD + 255) / 256;
  k0_fmaps<<<g0, 256, 0, stream>>>(kfmap, vfmap, kf, vfT);

  size_t totVT = (size_t)256 * HD * KPAD;
  int g1 = (int)((totVT + 255) / 256);
  k0_vt<<<g1, 256, 0, stream>>>(v, vT);

  dim3 grid(37, 256);
  k1_scores<<<grid, 256, 0, stream>>>(q, k, kgem, kf, attn_ptr);
  k2_out<<<grid, 256, 0, stream>>>(vgem, vfT, vT, attn_ptr, out_ptr);
}
